// DBRXAttention_23587960390188
// MI455X (gfx1250) — matrix operand order repaired and checked by reading, not compiled, not run on hardware
//
#include <hip/hip_runtime.h>
#include <math.h>
#include <stdint.h>

#define T_SEQ   2048
#define DMODEL  6144
#define NHEADS  48
#define NKV     8
#define HD      128
#define QKV_N   8192
#define GQA_REP 6

typedef __bf16 bf16;
typedef __attribute__((ext_vector_type(16))) __bf16 v16bf;
typedef __attribute__((ext_vector_type(8)))  __bf16 v8bf;
typedef __attribute__((ext_vector_type(8)))  float  v8f;
typedef __attribute__((ext_vector_type(4)))  int    v4i;

#if defined(__AMDGCN__) && __has_builtin(__builtin_amdgcn_global_load_async_to_lds_b128)
#define HAVE_ASYNC_LDS 1
#else
#define HAVE_ASYNC_LDS 0
#endif

typedef __attribute__((address_space(1))) v4i* gas1_v4ip;
typedef __attribute__((address_space(3))) v4i* as3_v4ip;

__device__ __forceinline__ void async_copy_b128(const bf16* gsrc, bf16* ldst) {
#if HAVE_ASYNC_LDS
  gas1_v4ip s = (gas1_v4ip)(unsigned long long)(uintptr_t)gsrc;
  as3_v4ip  d = (as3_v4ip)(uint32_t)(uintptr_t)ldst;   // low 32 bits = LDS offset
  __builtin_amdgcn_global_load_async_to_lds_b128(s, d, 0, 0);
#else
  *(v16bf*)ldst = *(const v16bf*)gsrc;                 // sync fallback
#endif
}

__device__ __forceinline__ void wait_async_all() {
#if HAVE_ASYNC_LDS
#if __has_builtin(__builtin_amdgcn_s_wait_asynccnt)
  __builtin_amdgcn_s_wait_asynccnt(0);
#else
  asm volatile("s_wait_asynccnt 0" ::: "memory");
#endif
#endif
}

__device__ __forceinline__ v8f wmma_bf16(v16bf a, v16bf b, v8f c) {
  return __builtin_amdgcn_wmma_f32_16x16x32_bf16(false, a, false, b, (short)0, c,
                                                 false, false);
}

// A-fragment (16x32, 16-bit) load. rowPtr -> A[row m] + k0.
// lanes 0-15 (g=0): halves j=0..15 -> K=j
// lanes 16-31(g=1): halves j=0..15 -> K=16+j
__device__ __forceinline__ v16bf load_afrag(const bf16* rowPtr, int g) {
  v8bf lo = *(const v8bf*)(rowPtr + g * 16);
  v8bf hi = *(const v8bf*)(rowPtr + g * 16 + 8);
  v16bf r;
#pragma unroll
  for (int j = 0; j < 8; ++j) { r[j] = lo[j]; r[j + 8] = hi[j]; }
  return r;
}

// ---------------------------------------------------------------------------
// fp32 -> bf16 bulk convert (x4 vectorized)
// ---------------------------------------------------------------------------
__global__ void cvt_f32_bf16_x4(const float* __restrict__ in,
                                bf16* __restrict__ out, int n4) {
  int i = blockIdx.x * blockDim.x + threadIdx.x;
  if (i >= n4) return;
  float4 f = ((const float4*)in)[i];
  out[4 * i + 0] = (bf16)f.x;
  out[4 * i + 1] = (bf16)f.y;
  out[4 * i + 2] = (bf16)f.z;
  out[4 * i + 3] = (bf16)f.w;
}

// ---------------------------------------------------------------------------
// C[M][N] (f32) = A[M][K] (bf16, row-major) * B[N][K]^T (bf16, row-major)
// 256 threads = 8 waves; tile 128x128; K step 32.
// Double-buffered LDS fed by GLOBAL_LOAD_ASYNC_TO_LDS_B128 (ASYNCcnt).
// ---------------------------------------------------------------------------
__global__ __launch_bounds__(256)
void gemm_nt_bf16(const bf16* __restrict__ A, const bf16* __restrict__ B,
                  float* __restrict__ C, int M, int N, int K) {
  __shared__ bf16 sA[2][128][40];   // 80B rows: 16B aligned, bank-spread
  __shared__ bf16 sB[2][128][40];
  const int tid  = threadIdx.x;
  const int lane = tid & 31;
  const int wid  = tid >> 5;
  const int wm   = wid & 3;          // 4 row-strips of 32
  const int wn   = wid >> 2;         // 2 col-strips of 64
  const int g    = (lane >> 4) & 1;
  const int hn   = lane & 15;
  const int bm   = blockIdx.y * 128;
  const int bn   = blockIdx.x * 128;
  const int lr   = tid >> 1;         // 0..127 : tile row this thread stages
  const int seg  = (tid & 1) * 16;   // half-row segment (16 halves = 32B)

  const bf16* srcA = A + (size_t)(bm + lr) * K + seg;
  const bf16* srcB = B + (size_t)(bn + lr) * K + seg;

  v8f zz = {};
  v8f acc[2][4];
#pragma unroll
  for (int mt = 0; mt < 2; ++mt)
#pragma unroll
    for (int nt = 0; nt < 4; ++nt) acc[mt][nt] = zz;

  const int nk = K / 32;
  // prologue: stage tile 0 into buffer 0
  async_copy_b128(srcA, &sA[0][lr][seg]);
  async_copy_b128(srcB, &sB[0][lr][seg]);

  int buf = 0;
  for (int it = 0; it < nk; ++it) {
    wait_async_all();        // this wave's DMA for buffer `buf` has landed
    __syncthreads();         // ... and everyone else's too

    if (it + 1 < nk) {       // prefetch next tile into the other buffer
      int k1 = (it + 1) * 32;
      async_copy_b128(srcA + k1, &sA[buf ^ 1][lr][seg]);
      async_copy_b128(srcB + k1, &sB[buf ^ 1][lr][seg]);
    }

    v16bf af[2], bfr[4];
#pragma unroll
    for (int mt = 0; mt < 2; ++mt)
      af[mt] = load_afrag(&sA[buf][wm * 32 + mt * 16 + hn][0], g);
#pragma unroll
    for (int nt = 0; nt < 4; ++nt)
      bfr[nt] = *(const v16bf*)&sB[buf][wn * 64 + nt * 16 + hn][g * 16];

#pragma unroll
    for (int mt = 0; mt < 2; ++mt)
#pragma unroll
      for (int nt = 0; nt < 4; ++nt)
        acc[mt][nt] = wmma_bf16(af[mt], bfr[nt], acc[mt][nt]);

    buf ^= 1;
  }

#pragma unroll
  for (int mt = 0; mt < 2; ++mt)
#pragma unroll
    for (int nt = 0; nt < 4; ++nt)
#pragma unroll
      for (int v = 0; v < 8; ++v) {
        int row = bm + wm * 32 + mt * 16 + g * 8 + v;
        int col = bn + wn * 64 + nt * 16 + hn;
        C[(size_t)row * N + col] = acc[mt][nt][v];
      }
}

// ---------------------------------------------------------------------------
// RoPE + clamp for Q: qkv f32 [T][8192] -> q bf16 [48][T][128], scale folded
// ---------------------------------------------------------------------------
__global__ void rope_q_kernel(const float* __restrict__ qkv, bf16* __restrict__ q) {
  int idx = blockIdx.x * blockDim.x + threadIdx.x;  // T*48*64
  if (idx >= T_SEQ * NHEADS * 64) return;
  int i = idx & 63;
  int h = (idx >> 6) % NHEADS;
  int t = idx / (64 * NHEADS);
  const float* src = qkv + (size_t)t * QKV_N + h * HD + 2 * i;
  float a = src[0], b = src[1];
  float inv = __powf(500000.0f, -(float)i / 64.0f);
  float sn, cs;
  __sincosf((float)t * inv, &sn, &cs);
  float a2 = a * cs - b * sn;
  float b2 = a * sn + b * cs;
  const float sc = 0.08838834764831845f;  // 1/sqrt(128)
  a2 = fminf(fmaxf(a2, -8.f), 8.f) * sc;
  b2 = fminf(fmaxf(b2, -8.f), 8.f) * sc;
  bf16* dst = q + ((size_t)h * T_SEQ + t) * HD + 2 * i;
  dst[0] = (bf16)a2;
  dst[1] = (bf16)b2;
}

// RoPE + clamp for K: -> k bf16 [8][T][128]
__global__ void rope_k_kernel(const float* __restrict__ qkv, bf16* __restrict__ k) {
  int idx = blockIdx.x * blockDim.x + threadIdx.x;  // T*8*64
  if (idx >= T_SEQ * NKV * 64) return;
  int i = idx & 63;
  int h = (idx >> 6) % NKV;
  int t = idx / (64 * NKV);
  const float* src = qkv + (size_t)t * QKV_N + DMODEL + h * HD + 2 * i;
  float a = src[0], b = src[1];
  float inv = __powf(500000.0f, -(float)i / 64.0f);
  float sn, cs;
  __sincosf((float)t * inv, &sn, &cs);
  float a2 = a * cs - b * sn;
  float b2 = a * sn + b * cs;
  a2 = fminf(fmaxf(a2, -8.f), 8.f);
  b2 = fminf(fmaxf(b2, -8.f), 8.f);
  bf16* dst = k + ((size_t)h * T_SEQ + t) * HD + 2 * i;
  dst[0] = (bf16)a2;
  dst[1] = (bf16)b2;
}

// clamp V and store transposed: -> vT bf16 [8][128][T]
__global__ void clamp_vT_kernel(const float* __restrict__ qkv, bf16* __restrict__ vT) {
  int idx = blockIdx.x * blockDim.x + threadIdx.x;  // T*8*128
  if (idx >= T_SEQ * NKV * HD) return;
  int d = idx & 127;
  int h = (idx >> 7) & 7;
  int t = idx >> 10;
  float v = qkv[(size_t)t * QKV_N + DMODEL + NKV * HD + h * HD + d];
  v = fminf(fmaxf(v, -8.f), 8.f);
  vT[((size_t)h * HD + d) * T_SEQ + t] = (bf16)v;
}

// ---------------------------------------------------------------------------
// Flash attention: grid (T/64, 48), block 128 (4 waves, 16 q-rows each).
// Q bf16 [48][T][128], K bf16 [8][T][128], VT bf16 [8][128][T]
// out: attn bf16 [T][6144]
// ---------------------------------------------------------------------------
__global__ __launch_bounds__(128)
void attn_fwd(const bf16* __restrict__ Q, const bf16* __restrict__ Kb,
              const bf16* __restrict__ VT, bf16* __restrict__ O) {
  __shared__ bf16 sP[4][16][40];   // per-wave C-layout -> A-layout staging
  const int tid  = threadIdx.x;
  const int lane = tid & 31;
  const int wid  = tid >> 5;
  const int g    = (lane >> 4) & 1;
  const int hn   = lane & 15;
  const int h    = blockIdx.y;
  const int kvh  = h / GQA_REP;
  const int qb   = blockIdx.x * 64 + wid * 16;

  const bf16* qh = Q  + (size_t)h   * T_SEQ * HD;
  const bf16* kh = Kb + (size_t)kvh * T_SEQ * HD;
  const bf16* vh = VT + (size_t)kvh * HD * T_SEQ;

  v16bf qf[4];
#pragma unroll
  for (int kc = 0; kc < 4; ++kc)
    qf[kc] = load_afrag(qh + (size_t)(qb + hn) * HD + kc * 32, g);

  v8f zz = {};
  v8f o[8];
#pragma unroll
  for (int i = 0; i < 8; ++i) o[i] = zz;
  float mrow[8], lrow[8];
#pragma unroll
  for (int v = 0; v < 8; ++v) { mrow[v] = -1e30f; lrow[v] = 0.f; }

  for (int kv0 = 0; kv0 <= qb + 15; kv0 += 32) {
    // ---- S = Q * K^T (two 16x16 tiles covering 32 keys) ----
    v8f s0 = zz, s1 = zz;
#pragma unroll
    for (int kc = 0; kc < 4; ++kc) {
      v16bf k0f = *(const v16bf*)(kh + (size_t)(kv0 + hn)      * HD + kc * 32 + g * 16);
      v16bf k1f = *(const v16bf*)(kh + (size_t)(kv0 + 16 + hn) * HD + kc * 32 + g * 16);
      s0 = wmma_bf16(qf[kc], k0f, s0);
      s1 = wmma_bf16(qf[kc], k1f, s1);
    }

    // ---- mask + online softmax (row = qb + g*8 + v, across 16 lanes) ----
    const int c0 = kv0 + hn, c1 = kv0 + 16 + hn;
    float p0[8], p1[8], alpha[8];
#pragma unroll
    for (int v = 0; v < 8; ++v) {
      int r = qb + g * 8 + v;
      float x0 = (c0 <= r) ? s0[v] : -1e30f;
      float x1 = (c1 <= r) ? s1[v] : -1e30f;
      float tm = fmaxf(x0, x1);
#pragma unroll
      for (int msk = 1; msk < 16; msk <<= 1)
        tm = fmaxf(tm, __shfl_xor(tm, msk, 32));
      float mn = fmaxf(mrow[v], tm);
      float a  = __expf(mrow[v] - mn);
      float e0 = __expf(x0 - mn);
      float e1 = __expf(x1 - mn);
      float rs = e0 + e1;
#pragma unroll
      for (int msk = 1; msk < 16; msk <<= 1)
        rs += __shfl_xor(rs, msk, 32);
      lrow[v] = lrow[v] * a + rs;
      mrow[v] = mn;
      alpha[v] = a;
      p0[v] = e0;
      p1[v] = e1;
    }
#pragma unroll
    for (int nc = 0; nc < 8; ++nc)
#pragma unroll
      for (int v = 0; v < 8; ++v) o[nc][v] *= alpha[v];

    // ---- P: C-layout -> A-layout via per-wave LDS ----
#pragma unroll
    for (int v = 0; v < 8; ++v) {
      sP[wid][g * 8 + v][hn]      = (bf16)p0[v];
      sP[wid][g * 8 + v][16 + hn] = (bf16)p1[v];
    }
    __builtin_amdgcn_wave_barrier();   // same-wave LDS ops are in-order (DScnt)
    v16bf pf = load_afrag(&sP[wid][hn][0], g);
    __builtin_amdgcn_wave_barrier();

    // ---- O += P * V (V transposed: contiguous along t) ----
#pragma unroll
    for (int nc = 0; nc < 8; ++nc) {
      v16bf vf = *(const v16bf*)(vh + (size_t)(nc * 16 + hn) * T_SEQ + kv0 + g * 16);
      o[nc] = wmma_bf16(pf, vf, o[nc]);
    }
  }

  // ---- epilogue: O /= l, store bf16 [t][h*128 + d] ----
#pragma unroll
  for (int nc = 0; nc < 8; ++nc)
#pragma unroll
    for (int v = 0; v < 8; ++v) {
      int r = qb + g * 8 + v;
      float val = o[nc][v] / lrow[v];
      O[(size_t)r * DMODEL + h * HD + nc * 16 + hn] = (bf16)val;
    }
}

// ---------------------------------------------------------------------------
extern "C" void kernel_launch(void* const* d_in, const int* in_sizes, int n_in,
                              void* d_out, int out_size, void* d_ws, size_t ws_size,
                              hipStream_t stream) {
  (void)in_sizes; (void)n_in; (void)out_size; (void)ws_size;
  const float* x    = (const float*)d_in[0];
  // d_in[1] = causal_mask (recomputed analytically in-kernel)
  const float* wqkv = (const float*)d_in[2];
  const float* wout = (const float*)d_in[3];
  float* out = (float*)d_out;

  char* ws = (char*)d_ws;
  bf16*  wqkv_b = (bf16*)ws;                 ws += (size_t)QKV_N  * DMODEL * 2;
  bf16*  wout_b = (bf16*)ws;                 ws += (size_t)DMODEL * DMODEL * 2;
  bf16*  x_b    = (bf16*)ws;                 ws += (size_t)T_SEQ  * DMODEL * 2;
  float* qkv_f  = (float*)ws;                ws += (size_t)T_SEQ  * QKV_N * 4;
  bf16*  q_b    = (bf16*)ws;                 ws += (size_t)NHEADS * T_SEQ * HD * 2;
  bf16*  k_b    = (bf16*)ws;                 ws += (size_t)NKV    * T_SEQ * HD * 2;
  bf16*  vT_b   = (bf16*)ws;                 ws += (size_t)NKV    * HD * T_SEQ * 2;
  bf16*  att_b  = (bf16*)ws;                 ws += (size_t)T_SEQ  * DMODEL * 2;

  // 1) bf16 conversions
  {
    int n4 = (QKV_N * DMODEL) / 4;
    cvt_f32_bf16_x4<<<(n4 + 255) / 256, 256, 0, stream>>>(wqkv, wqkv_b, n4);
    n4 = (DMODEL * DMODEL) / 4;
    cvt_f32_bf16_x4<<<(n4 + 255) / 256, 256, 0, stream>>>(wout, wout_b, n4);
    n4 = (T_SEQ * DMODEL) / 4;
    cvt_f32_bf16_x4<<<(n4 + 255) / 256, 256, 0, stream>>>(x, x_b, n4);
  }

  // 2) QKV projection: qkv[T][8192] = x[T][6144] * w_qkv[8192][6144]^T
  gemm_nt_bf16<<<dim3(QKV_N / 128, T_SEQ / 128), 256, 0, stream>>>(
      x_b, wqkv_b, qkv_f, T_SEQ, QKV_N, DMODEL);

  // 3) RoPE + clamp + layout shuffle
  rope_q_kernel<<<(T_SEQ * NHEADS * 64) / 256, 256, 0, stream>>>(qkv_f, q_b);
  rope_k_kernel<<<(T_SEQ * NKV * 64) / 256, 256, 0, stream>>>(qkv_f, k_b);
  clamp_vT_kernel<<<(T_SEQ * NKV * HD) / 256, 256, 0, stream>>>(qkv_f, vT_b);

  // 4) causal flash attention
  attn_fwd<<<dim3(T_SEQ / 64, NHEADS), 128, 0, stream>>>(q_b, k_b, vT_b, att_b);

  // 5) output projection: out[T][6144] = attn[T][6144] * w_out[6144][6144]^T
  gemm_nt_bf16<<<dim3(DMODEL / 128, T_SEQ / 128), 256, 0, stream>>>(
      att_b, wout_b, out, T_SEQ, DMODEL, DMODEL);
}
